// CueWordSelectNet_55886114455872
// MI455X (gfx1250) — compile-verified, matrix operand order checked by reading, not run on hardware
//
#include <hip/hip_runtime.h>
#include <hip/hip_bf16.h>

typedef unsigned short u16;
typedef __attribute__((ext_vector_type(16))) __bf16 v16bf;
typedef __attribute__((ext_vector_type(8)))  float  v8f;

#define B_SZ    256
#define HID     1000
#define NTOT    4000          // 4*HID gate width
#define T_STEPS 64
#define IN_SZ   600

// K (padded to multiple of 32) for each GEMM
#define KT1   82   // 2624 = 600+1000+1000 (+24 pad)
#define KT2   94   // 3008 = 1000+1000+1000 (+8 pad)
#define KTM   63   // 2016 = 1000+1000 (+16 pad)
#define KTM2  125  // 4000 exact

__device__ __forceinline__ u16 f32_to_bf16(float f) {
    unsigned int u = __float_as_uint(f);
    unsigned int r = u + 0x7FFFu + ((u >> 16) & 1u);   // round-to-nearest-even
    return (u16)(r >> 16);
}

// ---------------------------------------------------------------------------
// Generic fp32 -> bf16 packer into per-lane WMMA operand layout.
// Tile (rtile, ktile) = 32 lanes x 16 halves (1 KB), tileIdx = rtile*Ktiles+ktile.
// lane L: row r = rtile*16 + (L&15)
// half h: kk = (h<8) ? (L>>4)*8 + h : 16 + (L>>4)*8 + (h-8);  k = ktile*32+kk
// Source = up to 3 concatenated column segments (row stride ld, col offset off).
// ---------------------------------------------------------------------------
__global__ void pack_bf16(u16* __restrict__ dst,
                          const float* s0, int ld0, int off0, int len0,
                          const float* s1, int ld1, int off1, int len1,
                          const float* s2, int ld2, int off2, int len2,
                          int R, int Rtiles, int Ktiles) {
    long long idx = (long long)blockIdx.x * 256 + threadIdx.x;
    long long total = (long long)Rtiles * Ktiles * 512;
    if (idx >= total) return;
    int tile   = (int)(idx >> 9);
    int within = (int)(idx & 511);
    int lane = within >> 4;
    int h    = within & 15;
    int rtile = tile / Ktiles;
    int ktile = tile - rtile * Ktiles;
    int r  = rtile * 16 + (lane & 15);
    int kk = (h < 8) ? ((lane >> 4) * 8 + h) : (16 + (lane >> 4) * 8 + (h - 8));
    int k  = ktile * 32 + kk;
    float v = 0.0f;
    if (r < R) {
        if (k < len0)                       v = s0[(size_t)r * ld0 + off0 + k];
        else if (k < len0 + len1)           v = s1[(size_t)r * ld1 + off1 + (k - len0)];
        else if (k < len0 + len1 + len2)    v = s2[(size_t)r * ld2 + off2 + (k - len0 - len1)];
    }
    dst[idx] = f32_to_bf16(v);
}

// ---------------------------------------------------------------------------
// Async global -> LDS copy of one 1KB operand tile (32 lanes x 32B).
// ISA adds INST_OFFSET to BOTH the LDS and global address, so one address
// pair + offset:512 covers the second half. Tracked by ASYNCcnt.
// ---------------------------------------------------------------------------
__device__ __forceinline__ void async_copy_tile(const u16* gsrc, u16* lds_dst, int lane) {
    const char* g = (const char*)gsrc + lane * 16;
    unsigned l = (unsigned)(uintptr_t)lds_dst + lane * 16;
    asm volatile("global_load_async_to_lds_b128 %0, %1, off\n\t"
                 "global_load_async_to_lds_b128 %0, %1, off offset:512"
                 :: "v"(l), "v"(g) : "memory");
}

// ---------------------------------------------------------------------------
// C[256 x N] = A[256 x K] * W^T (+bias0+bias1), bf16 WMMA, f32 accumulate.
// grid.x = Ntiles (16 cols each). 256 threads = 8 waves; wave w owns M-tiles
// 2w, 2w+1 (rows 32w..32w+31).
// B tile staged once per block into LDS by wave 0 via async copy (double
// buffered, ASYNCcnt-pipelined); A tiles double-buffered in registers.
// ---------------------------------------------------------------------------
__global__ void __launch_bounds__(256)
gemm_wmma_bf16(const u16* __restrict__ Apack, const u16* __restrict__ Bpack,
               const float* __restrict__ bias0, const float* __restrict__ bias1,
               float* __restrict__ C, int ldc, int Ktiles, int Nlimit) {
    __shared__ u16 bsm[2][512];         // 2 x 1KB B-tile buffers
    const int ntile = blockIdx.x;
    const int wave  = threadIdx.x >> 5;
    const int lane  = threadIdx.x & 31;
    const int laneoff = lane * 16;

    const u16* Bt  = Bpack + (size_t)ntile * Ktiles * 512;
    const u16* At0 = Apack + (size_t)(wave * 2) * Ktiles * 512;
    const u16* At1 = At0 + (size_t)Ktiles * 512;

    // stage tile 0 into buffer 0
    if (wave == 0) async_copy_tile(Bt, &bsm[0][0], lane);

    v8f acc0 = {};
    v8f acc1 = {};

    // A register double buffer: preload tile 0
    v16bf a0 = *(const v16bf*)(At0 + laneoff);
    v16bf a1 = *(const v16bf*)(At1 + laneoff);

    for (int kt = 0; kt < Ktiles; ++kt) {
        const int cur = kt & 1;
        if (wave == 0) {
            if (kt + 1 < Ktiles) {
                // prefetch next B tile into the other buffer, then wait only
                // for the CURRENT tile (<=2 outstanding = next tile in flight)
                async_copy_tile(Bt + (size_t)(kt + 1) * 512, &bsm[cur ^ 1][0], lane);
                asm volatile("s_wait_asynccnt 0x2" ::: "memory");
            } else {
                asm volatile("s_wait_asynccnt 0x0" ::: "memory");
            }
        }
        __syncthreads();                 // B tile kt visible to all waves

        v16bf b = *(const v16bf*)(&bsm[cur][laneoff]);   // ds_load_b128 x2
        v16bf a0c = a0, a1c = a1;
        if (kt + 1 < Ktiles) {           // prefetch next A tiles into regs
            a0 = *(const v16bf*)(At0 + (size_t)(kt + 1) * 512 + laneoff);
            a1 = *(const v16bf*)(At1 + (size_t)(kt + 1) * 512 + laneoff);
        }
        acc0 = __builtin_amdgcn_wmma_f32_16x16x32_bf16(false, a0c, false, b,
                                                       (short)0, acc0, false, false);
        acc1 = __builtin_amdgcn_wmma_f32_16x16x32_bf16(false, a1c, false, b,
                                                       (short)0, acc1, false, false);
        __syncthreads();                 // reads done before buffer reuse
    }

    int n = ntile * 16 + (lane & 15);
    if (n < Nlimit) {
        float bias = 0.0f;
        if (bias0) bias += bias0[n];
        if (bias1) bias += bias1[n];
        // C/D layout: lane half selects M base (+0 / +8), VGPR e selects M+e
        int mbase = wave * 32 + ((lane >> 4) * 8);
        #pragma unroll
        for (int e = 0; e < 8; ++e) {
            C[(size_t)(mbase + e) * ldc + n]      = acc0[e] + bias;
            C[(size_t)(mbase + 16 + e) * ldc + n] = acc1[e] + bias;
        }
    }
}

// ---------------------------------------------------------------------------
// LSTM elementwise: gate order i, f, g, o in 4*HID-wide gates buffer.
// ---------------------------------------------------------------------------
__global__ void lstm_update(const float* __restrict__ gates,
                            float* __restrict__ h, float* __restrict__ c) {
    int idx = blockIdx.x * 256 + threadIdx.x;
    if (idx >= B_SZ * HID) return;
    int b = idx / HID;
    int j = idx - b * HID;
    const float* g = gates + (size_t)b * NTOT;
    float ig = 1.0f / (1.0f + __expf(-g[j]));
    float fg = 1.0f / (1.0f + __expf(-g[HID + j]));
    float gg = tanhf(g[2 * HID + j]);
    float og = 1.0f / (1.0f + __expf(-g[3 * HID + j]));
    float cn = fg * c[idx] + ig * gg;
    c[idx] = cn;
    h[idx] = og * tanhf(cn);
}

__global__ void fill_zero(float* __restrict__ p, int n) {
    int idx = blockIdx.x * 256 + threadIdx.x;
    if (idx < n) p[idx] = 0.0f;
}

__global__ void topic_scatter(const int* __restrict__ hidx, float* __restrict__ topic) {
    int idx = blockIdx.x * 256 + threadIdx.x;
    if (idx >= B_SZ * 5) return;
    int b = idx / 5;
    int t = hidx[idx];
    if (t >= 0 && t < HID) topic[(size_t)b * HID + t] = 1.0f;
}

// softmax over axis 0 (batch): one block per column j, 256 threads = 256 rows
__global__ void __launch_bounds__(256)
softmax_dim0(const float* __restrict__ m2, float* __restrict__ out) {
    __shared__ float red[256];
    int j = blockIdx.x;
    int tid = threadIdx.x;
    float v = m2[(size_t)tid * HID + j];
    red[tid] = v;
    __syncthreads();
    for (int s = 128; s > 0; s >>= 1) {
        if (tid < s) red[tid] = fmaxf(red[tid], red[tid + s]);
        __syncthreads();
    }
    float mx = red[0];
    __syncthreads();
    float e = __expf(v - mx);
    red[tid] = e;
    __syncthreads();
    for (int s = 128; s > 0; s >>= 1) {
        if (tid < s) red[tid] += red[tid + s];
        __syncthreads();
    }
    out[(size_t)tid * HID + j] = e / red[0];
}

__global__ void copy_states(const float* __restrict__ h1, const float* __restrict__ c1,
                            const float* __restrict__ h2, const float* __restrict__ c2,
                            float* __restrict__ out) {
    int idx = blockIdx.x * 256 + threadIdx.x;
    if (idx >= B_SZ * HID) return;
    out[256000 + idx]  = h1[idx];
    out[512000 + idx]  = c1[idx];
    out[768000 + idx]  = h2[idx];
    out[1024000 + idx] = c2[idx];
}

// ---------------------------------------------------------------------------
extern "C" void kernel_launch(void* const* d_in, const int* in_sizes, int n_in,
                              void* d_out, int out_size, void* d_ws, size_t ws_size,
                              hipStream_t stream) {
    const float* input = (const float*)d_in[0];   // (256, 64, 600)
    const int*   hcue  = (const int*)d_in[1];     // (256, 5)
    const float* W_ih1 = (const float*)d_in[2];   // (4000, 1600)
    const float* W_hh1 = (const float*)d_in[3];   // (4000, 1000)
    const float* b_ih1 = (const float*)d_in[4];
    const float* b_hh1 = (const float*)d_in[5];
    const float* W_ih2 = (const float*)d_in[6];   // (4000, 2600)
    const float* W_hh2 = (const float*)d_in[7];   // (4000, 1000)
    const float* b_ih2 = (const float*)d_in[8];
    const float* b_hh2 = (const float*)d_in[9];
    const float* W1    = (const float*)d_in[10];  // (4000, 2000)
    const float* b1    = (const float*)d_in[11];
    const float* W2    = (const float*)d_in[12];  // (1000, 4000)
    const float* b2    = (const float*)d_in[13];
    float* out = (float*)d_out;

    // ---- workspace carve-up (256B aligned) ----
    char* p = (char*)d_ws;
    auto alloc = [&](size_t bytes) -> void* {
        void* r = (void*)p;
        p += (bytes + 255) & ~(size_t)255;
        return r;
    };
    u16* Wc1  = (u16*)alloc((size_t)250 * KT1  * 512 * 2);  // [Wih1|Whh1]
    u16* Wc2  = (u16*)alloc((size_t)250 * KT2  * 512 * 2);  // [Wih2[:,600:2600]|Whh2]
    u16* W1p  = (u16*)alloc((size_t)250 * KTM  * 512 * 2);
    u16* W2p  = (u16*)alloc((size_t)63  * KTM2 * 512 * 2);
    u16* A1p  = (u16*)alloc((size_t)16 * KT1  * 512 * 2);
    u16* A2p  = (u16*)alloc((size_t)16 * KT2  * 512 * 2);
    u16* Amp  = (u16*)alloc((size_t)16 * KTM  * 512 * 2);
    u16* Am2p = (u16*)alloc((size_t)16 * KTM2 * 512 * 2);
    float* h1    = (float*)alloc(B_SZ * HID * 4);
    float* c1    = (float*)alloc(B_SZ * HID * 4);
    float* h2    = (float*)alloc(B_SZ * HID * 4);
    float* c2    = (float*)alloc(B_SZ * HID * 4);
    float* gates = (float*)alloc((size_t)B_SZ * NTOT * 4);
    float* topic = (float*)alloc(B_SZ * HID * 4);
    float* m1    = (float*)alloc((size_t)B_SZ * NTOT * 4);
    float* m2    = (float*)alloc(B_SZ * HID * 4);
    (void)ws_size; (void)n_in; (void)in_sizes; (void)out_size;

    auto packGrid = [](long long total) { return (int)((total + 255) / 256); };

    // ---- pack weights (bf16, WMMA operand layout), fused concatenations ----
    pack_bf16<<<packGrid((long long)250 * KT1 * 512), 256, 0, stream>>>(
        Wc1, W_ih1, 1600, 0, 1600, W_hh1, 1000, 0, 1000, nullptr, 0, 0, 0,
        4000, 250, KT1);
    pack_bf16<<<packGrid((long long)250 * KT2 * 512), 256, 0, stream>>>(
        Wc2, W_ih2, 2600, 600, 1000, W_ih2, 2600, 1600, 1000, W_hh2, 1000, 0, 1000,
        4000, 250, KT2);
    pack_bf16<<<packGrid((long long)250 * KTM * 512), 256, 0, stream>>>(
        W1p, W1, 2000, 0, 2000, nullptr, 0, 0, 0, nullptr, 0, 0, 0,
        4000, 250, KTM);
    pack_bf16<<<packGrid((long long)63 * KTM2 * 512), 256, 0, stream>>>(
        W2p, W2, 4000, 0, 4000, nullptr, 0, 0, 0, nullptr, 0, 0, 0,
        1000, 63, KTM2);

    // ---- zero initial states ----
    fill_zero<<<(B_SZ * HID + 255) / 256, 256, 0, stream>>>(h1, B_SZ * HID);
    fill_zero<<<(B_SZ * HID + 255) / 256, 256, 0, stream>>>(c1, B_SZ * HID);
    fill_zero<<<(B_SZ * HID + 255) / 256, 256, 0, stream>>>(h2, B_SZ * HID);
    fill_zero<<<(B_SZ * HID + 255) / 256, 256, 0, stream>>>(c2, B_SZ * HID);

    const int eGrid = (B_SZ * HID + 255) / 256;

    // ---- recurrent loop over T=64 timesteps ----
    for (int t = 0; t < T_STEPS; ++t) {
        // A1 = [x_t | h1 | h1]  (K = 600+1000+1000, padded to 2624)
        pack_bf16<<<packGrid((long long)16 * KT1 * 512), 256, 0, stream>>>(
            A1p, input, T_STEPS * IN_SZ, t * IN_SZ, IN_SZ,
            h1, HID, 0, HID, h1, HID, 0, HID, B_SZ, 16, KT1);
        gemm_wmma_bf16<<<250, 256, 0, stream>>>(A1p, Wc1, b_ih1, b_hh1,
                                                gates, NTOT, KT1, NTOT);
        lstm_update<<<eGrid, 256, 0, stream>>>(gates, h1, c1);

        // A2 = [h1 | h2 | h2]  (zeros-slice of W_ih2 skipped structurally)
        pack_bf16<<<packGrid((long long)16 * KT2 * 512), 256, 0, stream>>>(
            A2p, h1, HID, 0, HID, h2, HID, 0, HID, h2, HID, 0, HID, B_SZ, 16, KT2);
        gemm_wmma_bf16<<<250, 256, 0, stream>>>(A2p, Wc2, b_ih2, b_hh2,
                                                gates, NTOT, KT2, NTOT);
        lstm_update<<<eGrid, 256, 0, stream>>>(gates, h2, c2);
    }

    // ---- topic tracker: scatter 1.0 at cue-word ids ----
    fill_zero<<<eGrid, 256, 0, stream>>>(topic, B_SZ * HID);
    topic_scatter<<<(B_SZ * 5 + 255) / 256, 256, 0, stream>>>(hcue, topic);

    // ---- MLP head: m1 = [h1|topic] @ W1^T + b1 ; m2 = m1 @ W2^T + b2 ----
    pack_bf16<<<packGrid((long long)16 * KTM * 512), 256, 0, stream>>>(
        Amp, h1, HID, 0, HID, topic, HID, 0, HID, nullptr, 0, 0, 0, B_SZ, 16, KTM);
    gemm_wmma_bf16<<<250, 256, 0, stream>>>(Amp, W1p, b1, nullptr,
                                            m1, NTOT, KTM, NTOT);
    pack_bf16<<<packGrid((long long)16 * KTM2 * 512), 256, 0, stream>>>(
        Am2p, m1, NTOT, 0, NTOT, nullptr, 0, 0, 0, nullptr, 0, 0, 0, B_SZ, 16, KTM2);
    gemm_wmma_bf16<<<63, 256, 0, stream>>>(Am2p, W2p, b2, nullptr,
                                           m2, HID, KTM2, HID);

    // ---- softmax over batch axis, write outputs ----
    softmax_dim0<<<HID, 256, 0, stream>>>(m2, out);
    copy_states<<<eGrid, 256, 0, stream>>>(h1, c1, h2, c2, out);
}
